// HeteroGATLayer_31782757991162
// MI455X (gfx1250) — compile-verified, matrix operand order checked by reading, not run on hardware
//
#include <hip/hip_runtime.h>
#include <hip/hip_bf16.h>

typedef __attribute__((ext_vector_type(16))) __bf16 v16bf;
typedef __attribute__((ext_vector_type(8)))  __bf16 v8bf;
typedef __attribute__((ext_vector_type(8)))  float  v8f;

#define C128 128
#define NEG_SLOPE 0.2f

__device__ __forceinline__ float lrelu(float v) { return v > 0.f ? v : NEG_SLOPE * v; }

// float atomic max via signed/unsigned int atomics (bit-order trick)
__device__ __forceinline__ void atomicMaxFloat(float* addr, float v) {
    int iv = __float_as_int(v);
    if (iv >= 0) atomicMax((int*)addr, iv);
    else         atomicMin((unsigned int*)addr, (unsigned int)iv);
}

// ---------------- prep kernels ----------------
__global__ void prep_x_kernel(const float* __restrict__ x, const int* __restrict__ ntype,
                              const float* __restrict__ emb, __bf16* __restrict__ xh, int total) {
    int tid = blockIdx.x * blockDim.x + threadIdx.x;
    if (tid >= total) return;
    int i = tid >> 7, c = tid & 127;
    float v = x[tid] + emb[ntype[i] * C128 + c];
    xh[tid] = (__bf16)v;
}

// writes W transposed: wt[t][n][k] = W[t][k][n] (bf16), so GEMM staging is a straight copy
__global__ void prep_w_kernel(const float* __restrict__ W, __bf16* __restrict__ wt, int total) {
    int tid = blockIdx.x * blockDim.x + threadIdx.x;
    if (tid >= total) return;
    int t = tid >> 14;             // 128*128 = 16384
    int k = (tid >> 7) & 127;
    int n = tid & 127;
    wt[(t << 14) + (n << 7) + k] = (__bf16)W[tid];
}

// ---------------- GEMM: xw[N,128] = xh_bf16[N,128] @ W[t][128,128], f32 accum ----------------
// 256 threads = 8 waves; block covers 128 rows; each wave: 16 rows x 128 cols (8 tiles).
// Wt is already [n][k]; staged into LDS with 136-element row pitch via async load-to-LDS.
__global__ __launch_bounds__(256) void gemm_bf16_wmma(const __bf16* __restrict__ A,
                                                      const __bf16* __restrict__ Wt,
                                                      float* __restrict__ out, int N) {
    __shared__ __align__(16) __bf16 Bs[128 * 136];
    const int tid = threadIdx.x;

    // async global->LDS copy of Wt (32KB) in 16B chunks, inserting the 136 pitch
    for (int idx = tid; idx < 128 * 16; idx += 256) {
        int row = idx >> 4, chunk = idx & 15;
        unsigned long long gaddr = (unsigned long long)(Wt + row * 128 + chunk * 8);
        unsigned lds_off = (unsigned)(size_t)(const void*)&Bs[row * 136 + chunk * 8];
        asm volatile("global_load_async_to_lds_b128 %0, %1, off"
                     :: "v"(lds_off), "v"(gaddr) : "memory");
    }
    asm volatile("s_wait_asynccnt 0x0" ::: "memory");
    __syncthreads();

    const int lane  = tid & 31;
    const int wave  = tid >> 5;
    const int mbase = blockIdx.x * 128 + wave * 16;
    const int rlo   = lane & 15;          // A: row-in-tile; B: col-in-tile
    const int khalf = (lane >> 4) * 8;    // which 8-wide K slice this lane holds

    v8f acc[8] = {};

    #pragma unroll
    for (int kt = 0; kt < 4; ++kt) {
        const int kb = kt * 32;
        int arow = mbase + rlo;
        if (arow >= N) arow = N - 1;      // clamp: keep EXEC all-1s for WMMA
        const __bf16* ap = A + (size_t)arow * C128 + kb + khalf;
        v8bf alo = *(const v8bf*)ap;
        v8bf ahi = *(const v8bf*)(ap + 16);
        v16bf afrag = __builtin_shufflevector(alo, ahi, 0,1,2,3,4,5,6,7,8,9,10,11,12,13,14,15);

        // preload all 8 B fragments of this K step (loads clause together)
        v16bf bfrag[8];
        #pragma unroll
        for (int nt = 0; nt < 8; ++nt) {
            const __bf16* bp = &Bs[(nt * 16 + rlo) * 136 + kb + khalf];
            v8bf blo = *(const v8bf*)bp;
            v8bf bhi = *(const v8bf*)(bp + 16);
            bfrag[nt] = __builtin_shufflevector(blo, bhi, 0,1,2,3,4,5,6,7,8,9,10,11,12,13,14,15);
        }
        // 8 back-to-back WMMAs, independent accumulators
        #pragma unroll
        for (int nt = 0; nt < 8; ++nt) {
            acc[nt] = __builtin_amdgcn_wmma_f32_16x16x32_bf16(
                false, afrag, false, bfrag[nt], (short)0, acc[nt], false, false);
        }
    }

    const int n0 = lane & 15;
    const int mh = (lane >> 4) * 8;       // C/D layout: lanes 16-31 hold M = r+8
    #pragma unroll
    for (int nt = 0; nt < 8; ++nt) {
        #pragma unroll
        for (int r = 0; r < 8; ++r) {
            int row = mbase + mh + r;
            if (row < N) out[(size_t)row * C128 + nt * 16 + n0] = acc[nt][r];
        }
    }
}

// ---------------- per-node alpha + segment-max init ----------------
// wave-per-node: a_s = xw . att_src, a_d = xw . att_dst, m = leaky(a_s + a_d)
__global__ __launch_bounds__(256) void alpha_kernel(const float* __restrict__ xw,
                                                    const float* __restrict__ asrc,
                                                    const float* __restrict__ adst,
                                                    float* __restrict__ a_s, float* __restrict__ a_d,
                                                    float* __restrict__ m, int N) {
    int lane = threadIdx.x & 31;
    int node = blockIdx.x * 8 + (threadIdx.x >> 5);
    if (node >= N) return;
    float4 xv = ((const float4*)(xw + (size_t)node * C128))[lane];
    float4 sv = ((const float4*)asrc)[lane];
    float4 dv = ((const float4*)adst)[lane];
    float s = xv.x * sv.x + xv.y * sv.y + xv.z * sv.z + xv.w * sv.w;
    float d = xv.x * dv.x + xv.y * dv.y + xv.z * dv.z + xv.w * dv.w;
    #pragma unroll
    for (int off = 16; off > 0; off >>= 1) {
        s += __shfl_down(s, off, 32);
        d += __shfl_down(d, off, 32);
    }
    if (lane == 0) {
        a_s[node] = s;
        a_d[node] = d;
        m[node] = lrelu(s + d);
    }
}

// ---------------- edge segment-max ----------------
__global__ void edge_max_kernel(const int* __restrict__ src, const int* __restrict__ dst,
                                const int* __restrict__ etype, const float* __restrict__ a_s,
                                const float* __restrict__ a_d, float* __restrict__ m,
                                int E, int t) {
    int e = blockIdx.x * blockDim.x + threadIdx.x;
    if (e >= E || etype[e] != t) return;
    float logit = lrelu(a_s[src[e]] + a_d[dst[e]]);
    atomicMaxFloat(&m[dst[e]], logit);
}

// ---------------- self-loop init of wsum/num ----------------
__global__ void self_init_kernel(const float* __restrict__ xw, const float* __restrict__ a_s,
                                 const float* __restrict__ a_d, const float* __restrict__ m,
                                 float* __restrict__ wsum, float* __restrict__ num, int total) {
    int tid = blockIdx.x * blockDim.x + threadIdx.x;
    if (tid >= total) return;
    int i = tid >> 7, c = tid & 127;
    float e_self = lrelu(a_s[i] + a_d[i]);
    float w = __expf(e_self - m[i]);
    num[tid] = w * xw[tid];
    if (c == 0) wsum[i] = w;
}

// ---------------- edge weighted scatter (wave per edge) ----------------
__global__ __launch_bounds__(256) void edge_accum_kernel(const int* __restrict__ src,
                                                         const int* __restrict__ dst,
                                                         const int* __restrict__ etype,
                                                         const float* __restrict__ a_s,
                                                         const float* __restrict__ a_d,
                                                         const float* __restrict__ m,
                                                         const float* __restrict__ xw,
                                                         float* __restrict__ wsum,
                                                         float* __restrict__ num, int E, int t) {
    int lane = threadIdx.x & 31;
    int e = blockIdx.x * 8 + (threadIdx.x >> 5);
    if (e >= E) return;            // uniform per wave
    if (etype[e] != t) return;     // uniform per wave
    int s = src[e], d = dst[e];
    float w = __expf(lrelu(a_s[s] + a_d[d]) - m[d]);
    if (lane == 0) atomicAdd(&wsum[d], w);
    float4 xv = ((const float4*)(xw + (size_t)s * C128))[lane];
    float* np = num + (size_t)d * C128 + lane * 4;
    atomicAdd(np + 0, w * xv.x);
    atomicAdd(np + 1, w * xv.y);
    atomicAdd(np + 2, w * xv.z);
    atomicAdd(np + 3, w * xv.w);
}

// ---------------- finalize: out (=/+=) num/wsum + bias ----------------
__global__ void finalize_kernel(const float* __restrict__ num, const float* __restrict__ wsum,
                                const float* __restrict__ bias, float* __restrict__ out,
                                int total, int first) {
    int tid = blockIdx.x * blockDim.x + threadIdx.x;
    if (tid >= total) return;
    int i = tid >> 7, c = tid & 127;
    float v = num[tid] / wsum[i] + bias[c];
    out[tid] = first ? v : (out[tid] + v);
}

extern "C" void kernel_launch(void* const* d_in, const int* in_sizes, int n_in,
                              void* d_out, int out_size, void* d_ws, size_t ws_size,
                              hipStream_t stream) {
    const float* x       = (const float*)d_in[0];
    const int*   ei      = (const int*)  d_in[1];
    const int*   etype   = (const int*)  d_in[2];
    const int*   ntype   = (const int*)  d_in[3];
    const float* emb     = (const float*)d_in[4];
    const float* W       = (const float*)d_in[5];
    const float* att_src = (const float*)d_in[6];
    const float* att_dst = (const float*)d_in[7];
    const float* bias    = (const float*)d_in[8];
    float* out = (float*)d_out;

    const int N = in_sizes[0] / C128;
    const int E = in_sizes[2];
    const int* src = ei;
    const int* dst = ei + E;
    const int NC = N * C128;

    // workspace carve-out (256B aligned)
    char* ws = (char*)d_ws;
    size_t off = 0;
    auto alloc = [&](size_t bytes) -> void* {
        void* p = ws + off;
        off = (off + bytes + 255) & ~(size_t)255;
        return p;
    };
    __bf16* xh_bf = (__bf16*)alloc((size_t)NC * sizeof(__bf16));
    __bf16* wt_bf = (__bf16*)alloc((size_t)3 * C128 * C128 * sizeof(__bf16));
    float*  xw    = (float*) alloc((size_t)NC * sizeof(float));
    float*  num   = (float*) alloc((size_t)NC * sizeof(float));
    float*  a_s   = (float*) alloc((size_t)N * sizeof(float));
    float*  a_d   = (float*) alloc((size_t)N * sizeof(float));
    float*  m     = (float*) alloc((size_t)N * sizeof(float));
    float*  wsum  = (float*) alloc((size_t)N * sizeof(float));

    const int B256 = 256;
    prep_x_kernel<<<(NC + B256 - 1) / B256, B256, 0, stream>>>(x, ntype, emb, xh_bf, NC);
    prep_w_kernel<<<(3 * C128 * C128 + B256 - 1) / B256, B256, 0, stream>>>(W, wt_bf, 3 * C128 * C128);

    for (int t = 0; t < 3; ++t) {
        gemm_bf16_wmma<<<(N + 127) / 128, 256, 0, stream>>>(
            xh_bf, wt_bf + (size_t)t * C128 * C128, xw, N);
        alpha_kernel<<<(N + 7) / 8, 256, 0, stream>>>(
            xw, att_src + t * C128, att_dst + t * C128, a_s, a_d, m, N);
        edge_max_kernel<<<(E + B256 - 1) / B256, B256, 0, stream>>>(
            src, dst, etype, a_s, a_d, m, E, t);
        self_init_kernel<<<(NC + B256 - 1) / B256, B256, 0, stream>>>(
            xw, a_s, a_d, m, wsum, num, NC);
        edge_accum_kernel<<<(E + 7) / 8, 256, 0, stream>>>(
            src, dst, etype, a_s, a_d, m, xw, wsum, num, E, t);
        finalize_kernel<<<(NC + B256 - 1) / B256, B256, 0, stream>>>(
            num, wsum, bias + t * C128, out, NC, t == 0 ? 1 : 0);
    }
}
